// SynergyPre_77051713290672
// MI455X (gfx1250) — matrix operand order repaired and checked by reading, not compiled, not run on hardware
//
#include <hip/hip_runtime.h>
#include <math.h>

// ---------------------------------------------------------------------------
// SynergyPre on MI455X (gfx1250, wave32, WMMA).
//
// Reference notes:
//  * _cell_proj (cell_hidden_out @ proj_W.T + proj_b) is DEAD CODE in the
//    reference -- cell_fea gathers from drug_hidden_out. We skip it.
//  * Real work: per-edge triple gather + elementwise product (E x 128),
//    then MLP 128->32->32->32->1 (relu,relu,relu,sigmoid).
//  * Output tuple (x[E], label[E]) concatenated flat as float.
//
// Mapping: one wave32 = one 16-edge tile = M dim of 16x16x32 f16 WMMA tiles.
// ---------------------------------------------------------------------------

typedef __attribute__((ext_vector_type(16))) _Float16 v16h;
typedef __attribute__((ext_vector_type(8)))  _Float16 v8h;
typedef __attribute__((ext_vector_type(4)))  _Float16 v4h;
typedef __attribute__((ext_vector_type(8)))  float    v8f;

#define WPB    8      // waves per block
#define TILE_M 16     // edges per wave
#define DFEAT  128

__device__ __forceinline__ v16h cat16(v8h lo, v8h hi) {
  return __builtin_shufflevector(lo, hi, 0,1,2,3,4,5,6,7,8,9,10,11,12,13,14,15);
}

// A fragment (16-bit, 16xK row-major in LDS). Matches B fragment order:
// lanes 0-15 hold K = 0..15 of the 32-chunk, lanes 16-31 K = 16..31.
// `row` points at (row m, K-chunk base).
__device__ __forceinline__ v16h load_a_frag(const _Float16* row, int hi) {
  const int off = hi * 16;
  v8h lo = *(const v8h*)(row + off);        // K = off .. off+7
  v8h h2 = *(const v8h*)(row + off + 8);    // K = off+8 .. off+15
  return cat16(lo, h2);
}

// B fragment (16-bit, KxN): lane = N column; lanes 0-15 hold K=0..15 of the
// chunk, lanes 16-31 hold K=16..31. `p` points at W[n][kchunk + hi*16],
// 16 contiguous f16.
__device__ __forceinline__ v16h load_b_frag(const _Float16* p) {
  v8h lo = *(const v8h*)(p);
  v8h h2 = *(const v8h*)(p + 8);
  return cat16(lo, h2);
}

__global__ __launch_bounds__(256, 2)
void synergy_mlp_kernel(const float* __restrict__ dh,     // (N_DRUGS, 128)
                        const int*   __restrict__ edges,  // (E, 4) int32
                        const float* __restrict__ W1, const float* __restrict__ b1,
                        const float* __restrict__ W2, const float* __restrict__ b2,
                        const float* __restrict__ W3, const float* __restrict__ b3,
                        const float* __restrict__ W4, const float* __restrict__ b4,
                        float* __restrict__ out, int E)
{
  __shared__ _Float16 W1h[32 * DFEAT];   // 8 KB, W1[n][k] f16
  __shared__ _Float16 W2h[32 * 32];      // 2 KB
  __shared__ _Float16 W3h[32 * 32];      // 2 KB
  __shared__ float    W4f[32];
  __shared__ float    B1s[32], B2s[32], B3s[32];
  __shared__ _Float16 prod[WPB][TILE_M * DFEAT]; // 32 KB: per-wave product rows
  __shared__ _Float16 hA[WPB][TILE_M * 32];      // 8 KB: activations ping
  __shared__ _Float16 hB[WPB][TILE_M * 32];      // 8 KB: activations pong

  const int tid = threadIdx.x;

  // ---- Phase 0: stage weights (f32 -> f16) into LDS -----------------------
  for (int i = tid; i < 32 * DFEAT; i += 256) W1h[i] = (_Float16)W1[i];
  for (int i = tid; i < 32 * 32; i += 256) {
    W2h[i] = (_Float16)W2[i];
    W3h[i] = (_Float16)W3[i];
  }
  if (tid < 32) {
    W4f[tid] = W4[tid];
    B1s[tid] = b1[tid];
    B2s[tid] = b2[tid];
    B3s[tid] = b3[tid];
  }
  __syncthreads();

  const int wave = tid >> 5;
  const int lane = tid & 31;
  const int m    = lane & 15;   // my matrix row (edge within tile)
  const int hi   = lane >> 4;   // half-wave select

  const int base = blockIdx.x * (WPB * TILE_M) + wave * TILE_M;
  int eidx = base + m;
  eidx = (eidx < E) ? eidx : (E - 1);           // clamp (no EXEC divergence)
  const int4 e = ((const int4*)edges)[eidx];    // e.x,e.y,e.z = drug idx, e.w = label

  // ---- Phase 1: gather 3 rows, elementwise product -> prod (f16) ----------
  {
    const float4* r1 = (const float4*)(dh + (size_t)e.x * DFEAT);
    const float4* r2 = (const float4*)(dh + (size_t)e.y * DFEAT);
    const float4* r3 = (const float4*)(dh + (size_t)e.z * DFEAT);
    _Float16* prow = &prod[wave][m * DFEAT];
    const int j0 = hi * 16;                     // each half-wave does 64 feats
#pragma unroll
    for (int i = 0; i < 16; ++i) {
      float4 a = r1[j0 + i], b = r2[j0 + i], c = r3[j0 + i];
      v4h p;
      p[0] = (_Float16)(a.x * b.x * c.x);
      p[1] = (_Float16)(a.y * b.y * c.y);
      p[2] = (_Float16)(a.z * b.z * c.z);
      p[3] = (_Float16)(a.w * b.w * c.w);
      *(v4h*)(prow + (j0 + i) * 4) = p;
    }
  }
  __syncthreads();

  const int n0 = m;  // my N column within each 16-wide tile

  // ---- Layer 1: (16x128) @ (128x32) via 4 K-chunks x 2 N-tiles ------------
  v8f c0 = {};
  v8f c1 = {};
#pragma unroll
  for (int kc = 0; kc < 4; ++kc) {
    v16h a  = load_a_frag(&prod[wave][m * DFEAT + kc * 32], hi);
    v16h bt0 = load_b_frag(&W1h[n0 * DFEAT + kc * 32 + hi * 16]);
    c0 = __builtin_amdgcn_wmma_f32_16x16x32_f16(false, a, false, bt0,
                                                (short)0, c0, false, false);
    v16h bt1 = load_b_frag(&W1h[(16 + n0) * DFEAT + kc * 32 + hi * 16]);
    c1 = __builtin_amdgcn_wmma_f32_16x16x32_f16(false, a, false, bt1,
                                                (short)0, c1, false, false);
  }
  {
    const float bb0 = B1s[n0], bb1 = B1s[16 + n0];
#pragma unroll
    for (int r = 0; r < 8; ++r) {
      const int row = r + hi * 8;   // C/D layout: VGPR r holds M = r + 8*hi
      hA[wave][row * 32 + n0]      = (_Float16)fmaxf(c0[r] + bb0, 0.f);
      hA[wave][row * 32 + 16 + n0] = (_Float16)fmaxf(c1[r] + bb1, 0.f);
    }
  }
  __syncthreads();

  // ---- Layer 2: (16x32) @ (32x32), read hA -> write hB --------------------
  {
    v8f d0 = {};
    v8f d1 = {};
    v16h a   = load_a_frag(&hA[wave][m * 32], hi);
    v16h bt0 = load_b_frag(&W2h[n0 * 32 + hi * 16]);
    d0 = __builtin_amdgcn_wmma_f32_16x16x32_f16(false, a, false, bt0,
                                                (short)0, d0, false, false);
    v16h bt1 = load_b_frag(&W2h[(16 + n0) * 32 + hi * 16]);
    d1 = __builtin_amdgcn_wmma_f32_16x16x32_f16(false, a, false, bt1,
                                                (short)0, d1, false, false);
    const float bb0 = B2s[n0], bb1 = B2s[16 + n0];
#pragma unroll
    for (int r = 0; r < 8; ++r) {
      const int row = r + hi * 8;
      hB[wave][row * 32 + n0]      = (_Float16)fmaxf(d0[r] + bb0, 0.f);
      hB[wave][row * 32 + 16 + n0] = (_Float16)fmaxf(d1[r] + bb1, 0.f);
    }
  }
  __syncthreads();

  // ---- Layer 3: (16x32) @ (32x32), read hB -> write hA --------------------
  {
    v8f d0 = {};
    v8f d1 = {};
    v16h a   = load_a_frag(&hB[wave][m * 32], hi);
    v16h bt0 = load_b_frag(&W3h[n0 * 32 + hi * 16]);
    d0 = __builtin_amdgcn_wmma_f32_16x16x32_f16(false, a, false, bt0,
                                                (short)0, d0, false, false);
    v16h bt1 = load_b_frag(&W3h[(16 + n0) * 32 + hi * 16]);
    d1 = __builtin_amdgcn_wmma_f32_16x16x32_f16(false, a, false, bt1,
                                                (short)0, d1, false, false);
    const float bb0 = B3s[n0], bb1 = B3s[16 + n0];
#pragma unroll
    for (int r = 0; r < 8; ++r) {
      const int row = r + hi * 8;
      hA[wave][row * 32 + n0]      = (_Float16)fmaxf(d0[r] + bb0, 0.f);
      hA[wave][row * 32 + 16 + n0] = (_Float16)fmaxf(d1[r] + bb1, 0.f);
    }
  }
  __syncthreads();

  // ---- Layer 4: (16x32) @ (32x1) + sigmoid; emit x and label --------------
  if (hi == 0) {
    const int eo = base + m;
    if (eo < E) {
      float s = b4[0];
      const _Float16* hr = &hA[wave][m * 32];
#pragma unroll
      for (int k = 0; k < 32; ++k) s += (float)hr[k] * W4f[k];
      out[eo]     = 1.f / (1.f + __expf(-s));   // x
      out[E + eo] = (float)e.w;                 // label
    }
  }
}

extern "C" void kernel_launch(void* const* d_in, const int* in_sizes, int n_in,
                              void* d_out, int out_size, void* d_ws, size_t ws_size,
                              hipStream_t stream) {
  (void)n_in; (void)d_ws; (void)ws_size; (void)out_size;
  // setup_inputs order:
  // 0 drug_hidden_out, 1 cell_hidden_out (unused), 2 all_edges,
  // 3 proj_W (unused), 4 proj_b (unused),
  // 5 W1, 6 b1, 7 W2, 8 b2, 9 W3, 10 b3, 11 W4, 12 b4
  const float* dh    = (const float*)d_in[0];
  const int*   edges = (const int*)  d_in[2];
  const float* W1    = (const float*)d_in[5];
  const float* b1    = (const float*)d_in[6];
  const float* W2    = (const float*)d_in[7];
  const float* b2    = (const float*)d_in[8];
  const float* W3    = (const float*)d_in[9];
  const float* b3    = (const float*)d_in[10];
  const float* W4    = (const float*)d_in[11];
  const float* b4    = (const float*)d_in[12];
  float* out = (float*)d_out;

  const int E = in_sizes[2] / 4;                   // (E,4) flattened
  const int edges_per_block = WPB * TILE_M;        // 128
  const int blocks = (E + edges_per_block - 1) / edges_per_block;

  synergy_mlp_kernel<<<dim3(blocks), dim3(256), 0, stream>>>(
      dh, edges, W1, b1, W2, b2, W3, b3, W4, b4, out, E);
}
